// MDTCSML_80762565034288
// MI455X (gfx1250) — compile-verified, matrix operand order checked by reading, not run on hardware
//
#include <hip/hip_runtime.h>
#include <math.h>

// ---------------- problem constants ----------------
#define BATCH   16
#define LWAV    480000
#define NFFT    512
#define HOP     256
#define TFRAMES 1874                 // (480000-512)/256 + 1
#define FRAMES  (BATCH * TFRAMES)    // 29984, multiple of 16
#define NBINS   257
#define SBIN    288                  // padded bin count (mult of 32 for WMMA K)
#define BCOL    272                  // DFT bin tiles: 17 x 16
#define NMELS   64
#define RES     128
#define CHALF   64
#define VOCAB   410
#define VPADT   26                   // 416/16 N-tiles for FC
#define TP      1920                 // padded T (30 tiles of 64)
#define PI_D    3.14159265358979323846

// packed f16 weights layout in workspace (element offsets)
#define PW16TOT 135168               // 4096 (pre) + 16*8192
#define W2TOT   139264               // 17*8192
#define FCWTOT  53248                // 416*128
#define PKTOT   (PW16TOT + W2TOT + FCWTOT)

typedef __attribute__((ext_vector_type(16))) _Float16 v16h;
typedef __attribute__((ext_vector_type(8)))  _Float16 v8h;
typedef __attribute__((ext_vector_type(8)))  float    v8f;

// ---------------- WMMA helpers (wave32, 16x16x32 f16 -> f32) ----------------
__device__ __forceinline__ v8f wmma_f16(v16h a, v16h b, v8f c) {
  return __builtin_amdgcn_wmma_f32_16x16x32_f16(false, a, false, b, (short)0, c,
                                                false, false);
}

// Per-lane fragment = two contiguous 8-half runs: K = hi*8+0..7, 16+hi*8+0..7.
// `base` points at row_start + k0 (16B aligned). Two b128 loads, no packing.
__device__ __forceinline__ v16h frag_ld(const _Float16* base, int hi) {
  v8h lo = *(const v8h*)(base + (hi << 3));
  v8h hh = *(const v8h*)(base + 16 + (hi << 3));
  return __builtin_shufflevector(lo, hh, 0, 1, 2, 3, 4, 5, 6, 7, 8, 9, 10, 11,
                                 12, 13, 14, 15);
}

// ---------------- kernel 1: DFT basis (Hann folded, transposed) + melT ------
__global__ __launch_bounds__(256) void k_init(const float* __restrict__ mel,
                                              _Float16* __restrict__ bcosT,
                                              _Float16* __restrict__ bsinT,
                                              _Float16* __restrict__ melBT) {
  int tid = blockIdx.x * 256 + threadIdx.x;
  if (tid < BCOL * NFFT) {           // [bin][n] layout, K(=n) contiguous
    int bin = tid / NFFT, nn = tid % NFFT;
    if (bin < NBINS) {
      double w = 0.5 - 0.5 * cos(2.0 * PI_D * (double)nn / (double)(NFFT - 1));
      double ang = 2.0 * PI_D * (double)bin * (double)nn / (double)NFFT;
      bcosT[tid] = (_Float16)(float)(w * cos(ang));
      bsinT[tid] = (_Float16)(float)(w * sin(ang));
    } else {
      bcosT[tid] = (_Float16)0.f;
      bsinT[tid] = (_Float16)0.f;
    }
  }
  if (tid < NMELS * SBIN) {          // [mel][bin] layout, K(=bin) contiguous
    int n = tid / SBIN, k = tid % SBIN;
    melBT[tid] = (k < NBINS) ? (_Float16)mel[k * NMELS + n] : (_Float16)0.f;
  }
}

// ---------------- kernel 2: pack all GEMM weights to f16 once ---------------
struct PackPtrs {
  const float* pw[17];   // [0]=pre (64x64), [1..16] stacks (64x128)
  const float* w2[17];   // 128x64 each
  const float* fcw;      // 128x410
};
__global__ __launch_bounds__(256) void k_pack(PackPtrs P,
                                              _Float16* __restrict__ pk) {
  int tid = blockIdx.x * 256 + threadIdx.x;
  if (tid < PW16TOT) {
    int blk, idx;
    if (tid < 4096) { blk = 0; idx = tid; }
    else { blk = 1 + ((tid - 4096) >> 13); idx = (tid - 4096) & 8191; }
    pk[tid] = (_Float16)P.pw[blk][idx];
  } else if (tid < PW16TOT + W2TOT) {
    int r = tid - PW16TOT;
    pk[tid] = (_Float16)P.w2[r >> 13][r & 8191];
  } else if (tid < PKTOT) {
    int r = tid - PW16TOT - W2TOT;
    int n = r >> 7, k = r & 127;     // fc_w transposed: [vocab][c]
    pk[tid] = (n < VOCAB) ? (_Float16)P.fcw[k * VOCAB + n] : (_Float16)0.f;
  }
}

// ---------------- kernel 3: DFT magnitude via WMMA --------------------------
__global__ __launch_bounds__(256) void k_dft(const float* __restrict__ wav,
                                             const _Float16* __restrict__ bcosT,
                                             const _Float16* __restrict__ bsinT,
                                             float* __restrict__ mag) {
  int gid = blockIdx.x * 8 + (threadIdx.x >> 5);
  if (gid >= TFRAMES * 17) return;   // wave-uniform exit
  int ft = gid / 17, nt = (gid % 17) << 4;
  int lane = threadIdx.x & 31, hi = lane >> 4, l15 = lane & 15;
  int f = ft * 16 + l15;
  int b = f / TFRAMES, t = f % TFRAMES;
  const float* wp = wav + (size_t)b * LWAV + (size_t)t * HOP;
  int n = nt + l15;
  const _Float16* bcp = bcosT + (size_t)n * NFFT;
  const _Float16* bsp = bsinT + (size_t)n * NFFT;
  v8f cre = {}, cim = {};
  for (int k0 = 0; k0 < NFFT; k0 += 32) {
    const float* ap = wp + k0 + (hi << 3);
    float4 f0 = *(const float4*)(ap);
    float4 f1 = *(const float4*)(ap + 4);
    float4 f2 = *(const float4*)(ap + 16);
    float4 f3 = *(const float4*)(ap + 20);
    v16h a;
    a[0] = (_Float16)f0.x;  a[1] = (_Float16)f0.y;
    a[2] = (_Float16)f0.z;  a[3] = (_Float16)f0.w;
    a[4] = (_Float16)f1.x;  a[5] = (_Float16)f1.y;
    a[6] = (_Float16)f1.z;  a[7] = (_Float16)f1.w;
    a[8] = (_Float16)f2.x;  a[9] = (_Float16)f2.y;
    a[10] = (_Float16)f2.z; a[11] = (_Float16)f2.w;
    a[12] = (_Float16)f3.x; a[13] = (_Float16)f3.y;
    a[14] = (_Float16)f3.z; a[15] = (_Float16)f3.w;
    cre = wmma_f16(a, frag_ld(bcp + k0, hi), cre);
    cim = wmma_f16(a, frag_ld(bsp + k0, hi), cim);
  }
#pragma unroll
  for (int r = 0; r < 8; ++r) {
    int fr = ft * 16 + hi * 8 + r;
    float re = cre[r], im = cim[r];
    mag[(size_t)fr * SBIN + n] = __builtin_amdgcn_sqrtf(re * re + im * im);
  }
}

// ---------------- kernel 4: per-(batch,bin) EMA + normalize -> f16 ----------
__global__ __launch_bounds__(256) void k_ema(const float* __restrict__ alpha_p,
                                             const float* __restrict__ mag,
                                             _Float16* __restrict__ nmag) {
  int tid = blockIdx.x * 256 + threadIdx.x;
  if (tid >= BATCH * SBIN) return;
  int b = tid / SBIN, bin = tid % SBIN;
  _Float16* np = nmag + (size_t)b * TFRAMES * SBIN + bin;
  if (bin >= NBINS) {                // zero the K-pad columns
    for (int t = 0; t < TFRAMES; ++t) np[(size_t)t * SBIN] = (_Float16)0.f;
    return;
  }
  const float* mp = mag + (size_t)b * TFRAMES * SBIN + bin;
  float al = 1.f / (1.f + __expf(-alpha_p[bin]));
  float om = 1.f - al;
  float c = mp[0];                   // carry init = mag[:,0] -> avg[0]==mag[0]
  for (int t = 0; t < TFRAMES; ++t) {
    if (t + 16 < TFRAMES) __builtin_prefetch(mp + (size_t)(t + 16) * SBIN, 0, 1);
    float m = mp[(size_t)t * SBIN];
    c = al * c + om * m;
    np[(size_t)t * SBIN] = (_Float16)(m / (c + 1e-8f));
  }
}

// ---------------- kernel 5: mel projection -> x0 (B,64,TP) ------------------
__global__ __launch_bounds__(256) void k_mel(const _Float16* __restrict__ nmag,
                                             const _Float16* __restrict__ melBT,
                                             float* __restrict__ x0) {
  int gid = blockIdx.x * 8 + (threadIdx.x >> 5);
  if (gid >= TFRAMES * 4) return;
  int ft = gid >> 2, nt = (gid & 3) << 4;
  int lane = threadIdx.x & 31, hi = lane >> 4, l15 = lane & 15;
  const _Float16* ap = nmag + (size_t)(ft * 16 + l15) * SBIN;
  int n = nt + l15;
  const _Float16* bp = melBT + (size_t)n * SBIN;
  v8f acc = {};
  for (int k0 = 0; k0 < SBIN; k0 += 32)
    acc = wmma_f16(frag_ld(ap + k0, hi), frag_ld(bp + k0, hi), acc);
#pragma unroll
  for (int r = 0; r < 8; ++r) {
    int f = ft * 16 + hi * 8 + r;
    int bb = f / TFRAMES, t = f % TFRAMES;
    x0[((size_t)(bb * NMELS + n)) * TP + t] = acc[r] + 1e-6f;
  }
}

// ---------------- kernel 6: fused TCN block ---------------------------------
// depthwise(K=5,dil) -> WMMA pw(64xCin) -> PReLU(a1) -> WMMA w2(128x64)
// -> +b2 (+residual) -> PReLU(a2) (-> PReLU(post_a)) (-> pooled accumulate)
__global__ __launch_bounds__(256) void k_block(
    const float* __restrict__ xin, float* __restrict__ xout,
    const float* __restrict__ dw, const _Float16* __restrict__ pw16,
    const float* __restrict__ a1, const _Float16* __restrict__ w216,
    const float* __restrict__ b2, const float* __restrict__ a2,
    const float* __restrict__ post_a, const float* __restrict__ pool_w,
    int pool_s, int pool_init, float* __restrict__ pooled, int cin,
    int cshift, int dil) {
  // all rows K-major, padded to K+8 halves: 16B-aligned rows, bank stride
  // == 4 dwords mod 32 -> the 16 lanes' b128 reads tile all 64 banks.
  __shared__ __align__(16) _Float16 s_pw[CHALF * (RES + 8)];  // [m][k]
  __shared__ __align__(16) _Float16 s_w2[RES * (CHALF + 8)];  // [m][k]
  __shared__ __align__(16) _Float16 s_dwT[64 * (RES + 8)];    // [t][c]
  __shared__ __align__(16) _Float16 s_o1T[64 * (CHALF + 8)];  // [t][m]
  int tid = threadIdx.x;
  int b  = blockIdx.x / (TP / 64);
  int t0 = (blockIdx.x % (TP / 64)) * 64;
  const int ldk = cin + 8;           // 72 or 136
  const int ld2 = CHALF + 8;         // 72

  // stage packed f16 weights with b128 copies
  for (int i = tid; i < ((CHALF * cin) >> 3); i += 256) {
    int e = i << 3, m = e >> cshift, k = e & (cin - 1);
    *(v8h*)(s_pw + m * ldk + k) = *(const v8h*)(pw16 + e);
  }
  for (int i = tid; i < ((RES * CHALF) >> 3); i += 256) {
    int e = i << 3, m = e >> 6, k = e & 63;
    *(v8h*)(s_w2 + m * ld2 + k) = *(const v8h*)(w216 + e);
  }

  // depthwise conv, stored transposed [t][c] (contiguous LDS stores)
  int rf = 4 * dil;
  for (int i = tid; i < (cin << 6); i += 256) {
    int c = i & (cin - 1), tl = i >> cshift, t = t0 + tl;
    const float* xr = xin + ((size_t)(b * cin + c)) * TP;
    float acc = 0.f;
#pragma unroll
    for (int j = 0; j < 5; ++j) {
      int ts = t - rf + j * dil;
      acc += dw[c * 5 + j] * (ts >= 0 ? xr[ts] : 0.f);
    }
    s_dwT[tl * ldk + c] = (_Float16)acc;
  }
  __syncthreads();

  int wid = tid >> 5, lane = tid & 31, hi = lane >> 4, l15 = lane & 15;

  // GEMM1: (64 x cin) x (cin x 64) -> o1^T ; 16 tiles, 2 per wave
  for (int tt = wid * 2; tt < wid * 2 + 2; ++tt) {
    int mt = (tt >> 2) << 4, nt = (tt & 3) << 4;
    const _Float16* arow = s_pw + (mt + l15) * ldk;
    const _Float16* brow = s_dwT + (nt + l15) * ldk;
    v8f acc = {};
    for (int k0 = 0; k0 < cin; k0 += 32)
      acc = wmma_f16(frag_ld(arow + k0, hi), frag_ld(brow + k0, hi), acc);
#pragma unroll
    for (int r = 0; r < 8; ++r) {
      int m = mt + hi * 8 + r;
      float v = acc[r];
      float al = a1[m];
      v = v >= 0.f ? v : al * v;
      s_o1T[(nt + l15) * ld2 + m] = (_Float16)v;
    }
  }
  __syncthreads();

  // GEMM2: (128 x 64) x (64 x 64) ; 32 tiles, 4 per wave
  int residual = (cin == RES);
  for (int tt = wid * 4; tt < wid * 4 + 4; ++tt) {
    int mt = (tt >> 2) << 4, nt = (tt & 3) << 4;
    const _Float16* arow = s_w2 + (mt + l15) * ld2;
    const _Float16* brow = s_o1T + (nt + l15) * ld2;
    v8f acc = {};
    for (int k0 = 0; k0 < CHALF; k0 += 32)
      acc = wmma_f16(frag_ld(arow + k0, hi), frag_ld(brow + k0, hi), acc);
    int t = t0 + nt + l15;
#pragma unroll
    for (int r = 0; r < 8; ++r) {
      int m = mt + hi * 8 + r;
      size_t oi = ((size_t)(b * RES + m)) * TP + t;
      float v = acc[r] + b2[m];
      if (residual) v += xin[oi];
      float av = a2[m];
      v = v >= 0.f ? v : av * v;
      if (post_a) { float pv = post_a[m]; v = v >= 0.f ? v : pv * v; }
      xout[oi] = v;
      if (pool_s >= 0) {
        float ps = pool_w[pool_s];
        size_t pi = ((size_t)b * TP + t) * RES + m;   // pooled is [b][t][c]
        pooled[pi] = pool_init ? ps * v : pooled[pi] + ps * v;
      }
    }
  }
}

// ---------------- kernel 7: pooled PReLU + FC (128 -> 410) ------------------
__global__ __launch_bounds__(256) void k_fc(const float* __restrict__ pooled,
                                            const float* __restrict__ pool_a,
                                            const _Float16* __restrict__ fcwT,
                                            const float* __restrict__ fc_b,
                                            float* __restrict__ out) {
  int gid = blockIdx.x * 8 + (threadIdx.x >> 5);
  if (gid >= TFRAMES * VPADT) return;
  int ft = gid / VPADT, nt = (gid % VPADT) << 4;
  int lane = threadIdx.x & 31, hi = lane >> 4, l15 = lane & 15;
  float pa = pool_a[0];
  int fA = ft * 16 + l15;
  int bA = fA / TFRAMES, tA = fA % TFRAMES;
  const float* hp = pooled + ((size_t)bA * TP + tA) * RES;  // [b][t][c]
  int n = nt + l15;
  const _Float16* bp = fcwT + (size_t)n * RES;
  v8f acc = {};
  for (int k0 = 0; k0 < RES; k0 += 32) {
    const float* ap = hp + k0 + (hi << 3);
    float4 f0 = *(const float4*)(ap);
    float4 f1 = *(const float4*)(ap + 4);
    float4 f2 = *(const float4*)(ap + 16);
    float4 f3 = *(const float4*)(ap + 20);
    float e[16] = {f0.x, f0.y, f0.z, f0.w, f1.x, f1.y, f1.z, f1.w,
                   f2.x, f2.y, f2.z, f2.w, f3.x, f3.y, f3.z, f3.w};
    v16h a;
#pragma unroll
    for (int h = 0; h < 16; ++h) {
      float v = e[h];
      a[h] = (_Float16)(v >= 0.f ? v : pa * v);
    }
    acc = wmma_f16(a, frag_ld(bp + k0, hi), acc);
  }
  if (n < VOCAB) {
    float bb = fc_b[n];
#pragma unroll
    for (int r = 0; r < 8; ++r) {
      int f = ft * 16 + hi * 8 + r;
      out[(size_t)f * VOCAB + n] = acc[r] + bb;
    }
  }
}

// ---------------- host launcher ---------------------------------------------
extern "C" void kernel_launch(void* const* d_in, const int* in_sizes, int n_in,
                              void* d_out, int out_size, void* d_ws,
                              size_t ws_size, hipStream_t stream) {
  (void)in_sizes; (void)n_in; (void)out_size; (void)ws_size;
  // inputs in setup_inputs() dict insertion order, depth-first
  int i = 0;
  const float* wav   = (const float*)d_in[i++];
  const float* alpha = (const float*)d_in[i++];
  const float* mel   = (const float*)d_in[i++];
  struct Blk { const float *dw, *pw, *a1, *w2, *b2, *a2; };
  auto rdBlk = [&]() -> Blk {
    Blk bb;
    bb.dw = (const float*)d_in[i++]; bb.pw = (const float*)d_in[i++];
    bb.a1 = (const float*)d_in[i++]; bb.w2 = (const float*)d_in[i++];
    bb.b2 = (const float*)d_in[i++]; bb.a2 = (const float*)d_in[i++];
    return bb;
  };
  Blk pre = rdBlk();
  const float* prelu_a = (const float*)d_in[i++];
  Blk st[16];
  for (int k = 0; k < 16; ++k) st[k] = rdBlk();
  const float* pool_w = (const float*)d_in[i++];
  const float* pool_a = (const float*)d_in[i++];
  const float* fc_w   = (const float*)d_in[i++];
  const float* fc_b   = (const float*)d_in[i++];

  // workspace (~93 MB); xA aliases the dead mag buffer
  char* ws = (char*)d_ws;
  size_t off = 0;
  auto take = [&](size_t bytes) -> char* {
    char* p = ws + off;
    off = (off + bytes + 255) & ~(size_t)255;
    return p;
  };
  _Float16* bcosT = (_Float16*)take((size_t)BCOL * NFFT * 2);
  _Float16* bsinT = (_Float16*)take((size_t)BCOL * NFFT * 2);
  _Float16* melBT = (_Float16*)take((size_t)NMELS * SBIN * 2);
  _Float16* pk    = (_Float16*)take((size_t)PKTOT * 2);
  float*    magb  = (float*)   take((size_t)FRAMES * SBIN * 4);
  _Float16* nmag  = (_Float16*)take((size_t)FRAMES * SBIN * 2);
  float*    x0    = (float*)   take((size_t)BATCH * NMELS * TP * 4);
  float*    xB    = (float*)   take((size_t)BATCH * RES * TP * 4);
  float*    pooled = (float*)  take((size_t)BATCH * TP * RES * 4);
  float*    xA = magb;  // mag dead after k_ema; pre-block fully overwrites

  // weight pack + front-end tables
  PackPtrs P;
  P.pw[0] = pre.pw; P.w2[0] = pre.w2;
  for (int k = 0; k < 16; ++k) { P.pw[1 + k] = st[k].pw; P.w2[1 + k] = st[k].w2; }
  P.fcw = fc_w;
  k_init<<<(BCOL * NFFT + 255) / 256, 256, 0, stream>>>(mel, bcosT, bsinT, melBT);
  k_pack<<<(PKTOT + 255) / 256, 256, 0, stream>>>(P, pk);

  // front-end
  k_dft<<<(TFRAMES * 17 + 7) / 8, 256, 0, stream>>>(wav, bcosT, bsinT, magb);
  k_ema<<<(BATCH * SBIN + 255) / 256, 256, 0, stream>>>(alpha, magb, nmag);
  k_mel<<<(TFRAMES * 4 + 7) / 8, 256, 0, stream>>>(nmag, melBT, x0);

  // packed weight offsets
  auto pwOff = [](int blk) -> size_t { return blk == 0 ? 0 : 4096 + (size_t)(blk - 1) * 8192; };
  auto w2Off = [](int blk) -> size_t { return (size_t)PW16TOT + (size_t)blk * 8192; };
  const _Float16* fcwT = pk + PW16TOT + W2TOT;

  // TCN: pre-block (64 -> 128, no residual, extra PReLU(prelu_a))
  const int NBLK = BATCH * (TP / 64);
  k_block<<<NBLK, 256, 0, stream>>>(x0, xA, pre.dw, pk + pwOff(0), pre.a1,
                                    pk + w2Off(0), pre.b2, pre.a2, prelu_a,
                                    nullptr, -1, 0, pooled, NMELS, 6, 1);
  // 4 stacks x 4 blocks, dilations 1,2,4,8; pool after each stack's last block
  float* bufs[2] = {xA, xB};
  for (int k = 0; k < 16; ++k) {
    float* xi = bufs[k & 1];
    float* xo = bufs[(k & 1) ^ 1];
    int dil = 1 << (k & 3);
    int s = k >> 2;
    bool tail = ((k & 3) == 3);
    k_block<<<NBLK, 256, 0, stream>>>(xi, xo, st[k].dw, pk + pwOff(1 + k),
                                      st[k].a1, pk + w2Off(1 + k), st[k].b2,
                                      st[k].a2, nullptr,
                                      tail ? pool_w : nullptr, tail ? s : -1,
                                      (s == 0) ? 1 : 0, pooled, RES, 7, dil);
  }

  // classifier
  k_fc<<<(TFRAMES * VPADT + 7) / 8, 256, 0, stream>>>(pooled, pool_a, fcwT,
                                                      fc_b, (float*)d_out);
}
